// RecModel_20212116095665
// MI455X (gfx1250) — matrix operand order repaired and checked by reading, not compiled, not run on hardware
//
#include <hip/hip_runtime.h>
#include <hip/hip_bf16.h>

// ---------------------------------------------------------------------------
// Fused rec-model forward for MI455X (gfx1250, wave32).
//   x = concat(embs[224], hist[16], dense[1])  -> fc1(241->200)+relu
//     -> fc2(200->80)+relu -> fc3(80->2)
// Memory-bound on the embedding/history gathers (~72 MB @ 23.3 TB/s ~ 3.1us);
// matmuls (2.1 GFLOP) done with v_wmma_f32_16x16x32_bf16 so compute is free.
// ---------------------------------------------------------------------------

#define T_FEAT   14
#define L_VOCAB  100000
#define E_DIM    16
#define H_HIST   50
#define K1       241     // fc1 input dim
#define N1       200     // fc1 output dim
#define K1P      256     // K1 padded to multiple of 32
#define K2P      224     // fc2 K (200) padded
#define N2       80
#define K3P      96      // fc3 K (80) padded
#define ROWS     128     // batch rows per workgroup
#define NTHREADS 256     // 8 waves of 32

typedef __bf16 bf16;
typedef __attribute__((ext_vector_type(8)))  bf16  bfx8;
typedef __attribute__((ext_vector_type(16))) bf16  v16bf;
typedef __attribute__((ext_vector_type(8)))  float v8f;
typedef __attribute__((ext_vector_type(4)))  float fx4;

// A-fragment (16x32 bf16): lane l (l0=l&15, hi=l>>4) holds row
// m0+l0, K = [kb*32+hi*16, +16) contiguous in elems 0..15.
__device__ __forceinline__ v16bf load_afrag(const bf16* __restrict__ rowp, int kchunk) {
  union { v16bf v; bfx8 h[2]; } u;
  u.h[0] = *(const bfx8*)(rowp + kchunk);
  u.h[1] = *(const bfx8*)(rowp + kchunk + 8);
  return u.v;
}

// B-fragment (32x16 bf16): lane l holds column n0+(l&15), K = [kb*32+hi*16, +16)
// contiguous.  Weights are row-major (N,K) fp32 in global; convert to bf16.
template <int LDK, int KDIM, int NROWS, bool ALIGN16>
__device__ __forceinline__ v16bf load_wfrag(const float* __restrict__ Wm, int nrow, int kbase) {
  v16bf b;
  if (nrow < NROWS) {
    const float* wr = Wm + (size_t)nrow * LDK + kbase;
    if (ALIGN16 && (kbase + 16 <= KDIM)) {
      const fx4* p4 = (const fx4*)wr;            // LDK*4 % 16 == 0 -> 16B aligned
      #pragma unroll
      for (int q = 0; q < 4; ++q) {
        fx4 v = p4[q];
        #pragma unroll
        for (int j = 0; j < 4; ++j) b[4 * q + j] = (bf16)v[j];
      }
    } else {
      #pragma unroll
      for (int j = 0; j < 16; ++j) {
        float x = (kbase + j < KDIM) ? wr[j] : 0.0f;
        b[j] = (bf16)x;
      }
    }
  } else {
    #pragma unroll
    for (int j = 0; j < 16; ++j) b[j] = (bf16)0.0f;
  }
  return b;
}

__global__ __launch_bounds__(NTHREADS)
void recmodel_fused(const int*   __restrict__ sparse,   // (B,14)
                    const float* __restrict__ dense,    // (B,1)
                    const int*   __restrict__ hist,     // (B,50)
                    const float* __restrict__ W,        // (14,100001,16)
                    const float* __restrict__ fc1_w,    // (200,241)
                    const float* __restrict__ fc1_b,    // (200)
                    const float* __restrict__ fc2_w,    // (80,200)
                    const float* __restrict__ fc2_b,    // (80)
                    const float* __restrict__ fc3_w,    // (2,80)
                    const float* __restrict__ fc3_b,    // (2)
                    float*       __restrict__ out)      // (B,2)
{
  __shared__ bf16 Xs [ROWS][K1P];   // 64 KB: concat input, bf16, zero-padded
  __shared__ bf16 H1s[ROWS][K2P];   // 56 KB: relu(fc1) activations
  __shared__ bf16 H2s[ROWS][K3P];   // 24 KB: relu(fc2) activations

  const int tid  = threadIdx.x;
  const int row0 = blockIdx.x * ROWS;

  // ------------------- Phase 1: gathers into LDS (bf16) -------------------
  // Embeddings: X[r][t*16 .. t*16+15] = W[t][sparse[r,t]+1][:]
  for (int task = tid; task < ROWS * T_FEAT; task += NTHREADS) {
    int r = task % ROWS;
    int t = task / ROWS;
    int idx = sparse[(size_t)(row0 + r) * T_FEAT + t] + 1;
    const fx4* s4 = (const fx4*)(W + ((size_t)t * (L_VOCAB + 1) + (size_t)idx) * E_DIM);
    fx4 v0 = s4[0], v1 = s4[1], v2 = s4[2], v3 = s4[3];
    bfx8 lo, hh;
    #pragma unroll
    for (int j = 0; j < 4; ++j) {
      lo[j]     = (bf16)v0[j];  lo[4 + j] = (bf16)v1[j];
      hh[j]     = (bf16)v2[j];  hh[4 + j] = (bf16)v3[j];
    }
    *(bfx8*)&Xs[r][t * E_DIM]     = lo;
    *(bfx8*)&Xs[r][t * E_DIM + 8] = hh;
  }

  // History sum: X[r][224+e] = sum_j W[0][hist[r,j]+1][e]
  // (indices are >=1 so the W0[0]=0 masking in the reference never triggers)
  for (int task = tid; task < ROWS * E_DIM; task += NTHREADS) {
    int e = task & (E_DIM - 1);
    int r = task >> 4;
    const int* hr = hist + (size_t)(row0 + r) * H_HIST;
    float s = 0.0f;
    for (int j = 0; j < H_HIST; ++j) {
      int idx = hr[j] + 1;
      s += W[(size_t)idx * E_DIM + e];
    }
    Xs[r][T_FEAT * E_DIM + e] = (bf16)s;
  }

  // Dense feature at col 240; zero-pad cols 241..255.
  for (int r = tid; r < ROWS; r += NTHREADS) Xs[r][240] = (bf16)dense[row0 + r];
  for (int i = tid; i < ROWS * 15; i += NTHREADS) Xs[i / 15][241 + (i % 15)] = (bf16)0.0f;
  // Zero K-padding of the activation buffers (fc1 itself writes 0 in 200..207).
  for (int i = tid; i < ROWS * 16; i += NTHREADS) H1s[i / 16][208 + (i % 16)] = (bf16)0.0f;
  for (int i = tid; i < ROWS * 16; i += NTHREADS) H2s[i / 16][80  + (i % 16)] = (bf16)0.0f;

  __syncthreads();

  // Wave/lane geometry: wave w owns batch-row tile [m0, m0+16).
  const int wave = tid >> 5;
  const int lane = tid & 31;
  const int l0   = lane & 15;
  const int hi   = lane >> 4;
  const int m0   = wave * 16;

  // ------------------- Phase 2: fc1 (241->200) + ReLU ---------------------
  {
    v16bf a[8];
    const bf16* xr = &Xs[m0 + l0][0];
    #pragma unroll
    for (int kb = 0; kb < 8; ++kb) a[kb] = load_afrag(xr, kb * 32 + hi * 16);
    for (int n = 0; n < 13; ++n) {           // 13*16 = 208 >= 200 output cols
      v8f c = {};
      #pragma unroll
      for (int kb = 0; kb < 8; ++kb) {
        v16bf b = load_wfrag<K1, K1, N1, false>(fc1_w, n * 16 + l0, kb * 32 + hi * 16);
        c = __builtin_amdgcn_wmma_f32_16x16x32_bf16(false, a[kb], false, b,
                                                    (short)0, c, false, false);
      }
      int  ncol = n * 16 + l0;
      float bias = (ncol < N1) ? fc1_b[ncol] : 0.0f;
      #pragma unroll
      for (int v = 0; v < 8; ++v) {          // C layout: (M = v + hi*8, N = l0)
        float y = c[v] + bias;
        H1s[m0 + hi * 8 + v][ncol] = (bf16)(y > 0.0f ? y : 0.0f);
      }
    }
  }
  __syncthreads();

  // ------------------- Phase 3: fc2 (200->80) + ReLU ----------------------
  {
    v16bf a[7];
    const bf16* xr = &H1s[m0 + l0][0];
    #pragma unroll
    for (int kb = 0; kb < 7; ++kb) a[kb] = load_afrag(xr, kb * 32 + hi * 16);
    for (int n = 0; n < 5; ++n) {            // 5*16 = 80 exact
      v8f c = {};
      #pragma unroll
      for (int kb = 0; kb < 7; ++kb) {
        v16bf b = load_wfrag<200, 200, N2, true>(fc2_w, n * 16 + l0, kb * 32 + hi * 16);
        c = __builtin_amdgcn_wmma_f32_16x16x32_bf16(false, a[kb], false, b,
                                                    (short)0, c, false, false);
      }
      int   ncol = n * 16 + l0;
      float bias = fc2_b[ncol];
      #pragma unroll
      for (int v = 0; v < 8; ++v) {
        float y = c[v] + bias;
        H2s[m0 + hi * 8 + v][ncol] = (bf16)(y > 0.0f ? y : 0.0f);
      }
    }
  }
  __syncthreads();

  // ------------------- Phase 4: fc3 (80->2), write output -----------------
  {
    v16bf a[3];
    const bf16* xr = &H2s[m0 + l0][0];
    #pragma unroll
    for (int kb = 0; kb < 3; ++kb) a[kb] = load_afrag(xr, kb * 32 + hi * 16);
    v8f c = {};
    #pragma unroll
    for (int kb = 0; kb < 3; ++kb) {
      v16bf b = load_wfrag<80, 80, 2, true>(fc3_w, l0, kb * 32 + hi * 16);
      c = __builtin_amdgcn_wmma_f32_16x16x32_bf16(false, a[kb], false, b,
                                                  (short)0, c, false, false);
    }
    if (l0 < 2) {
      float bias = fc3_b[l0];
      #pragma unroll
      for (int v = 0; v < 8; ++v) {
        int r = row0 + m0 + hi * 8 + v;
        out[(size_t)r * 2 + l0] = c[v] + bias;
      }
    }
  }
}

extern "C" void kernel_launch(void* const* d_in, const int* in_sizes, int n_in,
                              void* d_out, int out_size, void* d_ws, size_t ws_size,
                              hipStream_t stream) {
  const int*   sparse = (const int*)  d_in[0];
  const float* dense  = (const float*)d_in[1];
  const int*   hist   = (const int*)  d_in[2];
  const float* W      = (const float*)d_in[3];
  const float* fc1_w  = (const float*)d_in[4];
  const float* fc1_b  = (const float*)d_in[5];
  const float* fc2_w  = (const float*)d_in[6];
  const float* fc2_b  = (const float*)d_in[7];
  const float* fc3_w  = (const float*)d_in[8];
  const float* fc3_b  = (const float*)d_in[9];
  float* out = (float*)d_out;

  const int B = in_sizes[0] / T_FEAT;        // 16384
  dim3 grid(B / ROWS), block(NTHREADS);
  recmodel_fused<<<grid, block, 0, stream>>>(sparse, dense, hist, W,
                                             fc1_w, fc1_b, fc2_w, fc2_b,
                                             fc3_w, fc3_b, out);
}
